// LGCN_REL_EMB_70368744178405
// MI455X (gfx1250) — matrix operand order repaired and checked by reading, not compiled, not run on hardware
//
#include <hip/hip_runtime.h>

// CDNA5 / gfx1250, wave32. WMMA vector types per CDNA5_HIP.md.
typedef __attribute__((ext_vector_type(16))) _Float16 v16h;
typedef __attribute__((ext_vector_type(8)))  float    v8f;

// Reference: fr/to/rel: T int32; w1:(R,N,32); w2:(R,32,C); out:(N,C).
// E hard-coded to 32 (== WMMA K of v_wmma_f32_16x16x32_f16); C <= 64.
#define CPAD 64   // padded output columns (branch-free scatter)

// ---------------------------------------------------------------- init
__global__ void init_kernel(int* __restrict__ cnt, int RN,
                            float* __restrict__ h, int NE,
                            float* __restrict__ out_pad, int NP,
                            int* __restrict__ sorted, int TP,
                            int* __restrict__ hist, int R) {
  const int stride = gridDim.x * blockDim.x;
  const int tid = blockIdx.x * blockDim.x + threadIdx.x;
  for (int i = tid; i < NP; i += stride) out_pad[i] = 0.0f;
  for (int i = tid; i < NE; i += stride) h[i] = 0.0f;
  for (int i = tid; i < RN; i += stride) cnt[i] = 0;
  for (int i = tid; i < TP; i += stride) sorted[i] = 0;   // sentinel triple 0
  for (int i = tid; i < R;  i += stride) hist[i] = 0;
}

// ----------------------------------------------- per-(rel,src) counts + histogram
__global__ void count_kernel(const int* __restrict__ fr, const int* __restrict__ rel,
                             int* __restrict__ cnt, int* __restrict__ hist,
                             int N, int T) {
  const int t = blockIdx.x * blockDim.x + threadIdx.x;
  if (t < T) {
    atomicAdd(&cnt[rel[t] * N + fr[t]], 1);
    atomicAdd(&hist[rel[t]], 1);
  }
}

// ---------------------------------------- tiny prefix over R buckets (1 thread)
// Padded layout: segment r occupies sorted[tile_base[r]*32 .. tile_base[r+1]*32),
// real entries first (pend[r] = start + hist[r]), sentinel 0 in the padding.
__global__ void prefix_kernel(const int* __restrict__ hist,
                              int* __restrict__ pend, int* __restrict__ tile_base,
                              int* __restrict__ cursor, int R) {
  if (blockIdx.x == 0 && threadIdx.x == 0) {
    int toff = 0;
    for (int r = 0; r < R; ++r) {
      tile_base[r] = toff;
      cursor[r]    = toff * 32;
      pend[r]      = toff * 32 + hist[r];
      toff += (hist[r] + 31) >> 5;
    }
    tile_base[R] = toff;
  }
}

// ----------------------------------------------- counting-sort scatter by relation
__global__ void scatter_kernel(const int* __restrict__ rel, int* __restrict__ cursor,
                               int* __restrict__ sorted, int T) {
  const int t = blockIdx.x * blockDim.x + threadIdx.x;
  if (t < T) {
    const int pos = atomicAdd(&cursor[rel[t]], 1);
    sorted[pos] = t;
  }
}

// ------------------------------------------------------------------ edge norms
__global__ void norm_kernel(const int* __restrict__ fr, const int* __restrict__ rel,
                            const int* __restrict__ cnt, float* __restrict__ nrm,
                            int N, int T) {
  const int t = blockIdx.x * blockDim.x + threadIdx.x;
  if (t < T) {
    const int c = cnt[rel[t] * N + fr[t]];   // >= 1 (triple t contributes)
    nrm[t] = 1.0f / (float)c;
  }
}

// ------------- SpMM1 (float4): h[fr[t], q*4..q*4+3] += nrm[t] * w1[rel[t],to[t],..]
__global__ void spmm1_kernel(const int* __restrict__ fr, const int* __restrict__ to,
                             const int* __restrict__ rel,
                             const float* __restrict__ w1, const float* __restrict__ nrm,
                             float* __restrict__ h, int N, int T) {
  const int i = blockIdx.x * blockDim.x + threadIdx.x;
  if (i >= T * 8) return;
  const int t = i >> 3, q = i & 7;
  const float s = nrm[t];
  const float4 w = ((const float4*)(w1 + ((size_t)rel[t] * N + to[t]) * 32))[q];
  float* hp = h + (size_t)fr[t] * 32 + q * 4;
  atomicAdd(hp + 0, s * w.x);
  atomicAdd(hp + 1, s * w.y);
  atomicAdd(hp + 2, s * w.z);
  atomicAdd(hp + 3, s * w.w);
}

// --------------------------------------------------------------- h = relu(h + b1)
__global__ void relu_kernel(float* __restrict__ h, const float* __restrict__ bias1, int NE) {
  const int i = blockIdx.x * blockDim.x + threadIdx.x;
  if (i < NE) h[i] = fmaxf(h[i] + bias1[i & 31], 0.0f);
}

// ----------------------------------------------------------------- pack B (64 KB)
// Fragment-order f16 copy of w2: packedB[(((r*4)+nt)*32 + lane)*16 + j], where
// col = nt*16 + (lane&15), K = (lane>>4)*16 + j  (contiguous K order, same as A).
__global__ void pack_b_kernel(const float* __restrict__ w2, _Float16* __restrict__ packedB,
                              int R, int C) {
  const int i = blockIdx.x * blockDim.x + threadIdx.x;
  if (i >= R * 4 * 32) return;
  const int lane = i & 31;
  const int nt   = (i >> 5) & 3;
  const int r    = i >> 7;
  const int col  = nt * 16 + (lane & 15);
  const int kb   = (lane >> 4) * 16;
  _Float16* dst = packedB + (size_t)i * 16;
#pragma unroll
  for (int j = 0; j < 16; ++j)
    dst[j] = (col < C) ? (_Float16)w2[((size_t)r * 32 + kb + j) * C + col] : (_Float16)0.0f;
}

// ---------------------------------------------------------------------------------
// Fused stage: per relation r (triples sorted+padded by rel), GEMM
// (T_r x 32) @ (32 x C) via v_wmma_f32_16x16x32_f16; nrm folded into A;
// branch-free atomic scatter into 64-column padded output.
// One wave per 32-triple supertile (2 M-tiles x 4 N-tiles = 8 WMMAs).
// ---------------------------------------------------------------------------------
__global__ __launch_bounds__(256)
void wmma_out_kernel(const int* __restrict__ sorted,
                     const int* __restrict__ pend,
                     const int* __restrict__ tile_base,
                     const int* __restrict__ fr,
                     const int* __restrict__ to,
                     const float* __restrict__ nrm,
                     const float* __restrict__ h,
                     const _Float16* __restrict__ packedB,
                     float* __restrict__ out_pad,
                     int R) {
  const int lane = threadIdx.x & 31;
  const int tile = blockIdx.x * (blockDim.x >> 5) + (threadIdx.x >> 5);
  const int totalTiles = tile_base[R];
  if (tile >= totalTiles) return;   // wave-uniform exit (EXEC all-1s at WMMA)

  int r = 0;
  while (r + 1 < R && tile_base[r + 1] <= tile) ++r;
  const int p      = tile * 32;     // padded layout: global pos == tile*32
  const int segEnd = pend[r];       // first padded (invalid) position

  const int row   = lane & 15;      // M row within a 16-row tile
  const int khalf = lane >> 4;      // which K half this lane holds
  const int kq0   = khalf ? 4 : 0;  // float4 index of K {0..7} vs {16..23}
  const int kq1   = khalf ? 6 : 2;  // float4 index of K {8..15} vs {24..31}

  // ---- A fragments: rows of h gathered via b128, pre-scaled by nrm, f32->f16.
  // Padded rows stay zero -> their acc rows are exactly 0 -> scatter is a no-op.
  v16h A[2];
#pragma unroll
  for (int mt = 0; mt < 2; ++mt) {
    A[mt] = (v16h){};
    const int pos = p + mt * 16 + row;
    if (pos < segEnd) {
      const int tA = sorted[pos];
      const float s = nrm[tA];
      const float4* hr = (const float4*)(h + (size_t)to[tA] * 32);
      const float4 a0 = hr[kq0], a1 = hr[kq0 + 1];
      const float4 a2 = hr[kq1], a3 = hr[kq1 + 1];
      A[mt][0]  = (_Float16)(s * a0.x); A[mt][1]  = (_Float16)(s * a0.y);
      A[mt][2]  = (_Float16)(s * a0.z); A[mt][3]  = (_Float16)(s * a0.w);
      A[mt][4]  = (_Float16)(s * a1.x); A[mt][5]  = (_Float16)(s * a1.y);
      A[mt][6]  = (_Float16)(s * a1.z); A[mt][7]  = (_Float16)(s * a1.w);
      A[mt][8]  = (_Float16)(s * a2.x); A[mt][9]  = (_Float16)(s * a2.y);
      A[mt][10] = (_Float16)(s * a2.z); A[mt][11] = (_Float16)(s * a2.w);
      A[mt][12] = (_Float16)(s * a3.x); A[mt][13] = (_Float16)(s * a3.y);
      A[mt][14] = (_Float16)(s * a3.z); A[mt][15] = (_Float16)(s * a3.w);
    }
  }

  // ---- B fragments from packed buffer (1 contiguous 32B load per nt) + WMMAs
  const v16h* pb = (const v16h*)packedB + (size_t)r * 4 * 32;
  v8f acc[2][4];
#pragma unroll
  for (int mt = 0; mt < 2; ++mt)
#pragma unroll
    for (int nt = 0; nt < 4; ++nt) acc[mt][nt] = (v8f){};

#pragma unroll
  for (int nt = 0; nt < 4; ++nt) {
    const v16h B = pb[nt * 32 + lane];
#pragma unroll
    for (int mt = 0; mt < 2; ++mt) {
      // (neg_a, A, neg_b, B, c_mod, C, reuse_a, reuse_b)
      acc[mt][nt] = __builtin_amdgcn_wmma_f32_16x16x32_f16(
          false, A[mt], false, B, (short)0, acc[mt][nt], false, false);
    }
  }

  // ---- Branch-free scatter. D layout: VGPR i -> M = khalf*8 + i,
  // N = nt*16 + (lane&15). nt becomes an immediate byte offset.
  const int col = lane & 15;
#pragma unroll
  for (int mt = 0; mt < 2; ++mt) {
#pragma unroll
    for (int i = 0; i < 8; ++i) {
      const int pos = p + mt * 16 + khalf * 8 + i;
      const int tm  = sorted[pos];                      // padding -> sentinel 0
      float* orow = out_pad + (size_t)fr[tm] * CPAD + col;
#pragma unroll
      for (int nt = 0; nt < 4; ++nt)
        atomicAdd(orow + nt * 16, acc[mt][nt][i]);      // +0.0 for padded rows
    }
  }
}

// ------------------------------------------------- out = out_pad[:, :C] + bias2
__global__ void finalize_kernel(const float* __restrict__ out_pad,
                                const float* __restrict__ bias2,
                                float* __restrict__ out, int NC, int C) {
  const int i = blockIdx.x * blockDim.x + threadIdx.x;
  if (i < NC) {
    const int v = i / C, c = i - v * C;
    out[i] = out_pad[(size_t)v * CPAD + c] + bias2[c];
  }
}

// ---------------------------------------------------------------------------------
extern "C" void kernel_launch(void* const* d_in, const int* in_sizes, int n_in,
                              void* d_out, int out_size, void* d_ws, size_t ws_size,
                              hipStream_t stream) {
  const int*   fr    = (const int*)  d_in[0];
  const int*   to    = (const int*)  d_in[1];
  const int*   rel   = (const int*)  d_in[2];
  const float* w1    = (const float*)d_in[3];
  const float* w2    = (const float*)d_in[4];
  const float* bias1 = (const float*)d_in[5];
  const float* bias2 = (const float*)d_in[6];
  float*       out   = (float*)      d_out;

  const int T = in_sizes[0];
  const int E = in_sizes[5];                         // 32 (== WMMA K)
  const int C = in_sizes[6];                         // 50 (<= 64)
  const int N = out_size / C;
  const int R = (int)((long long)in_sizes[3] / ((long long)N * E));

  const int RN = R * N, NE = N * 32, NC = N * C;
  const int NP = N * CPAD;                           // padded output elems
  const int TP = T + 32 * (R + 1);                   // padded sorted capacity

  // ---- workspace carve (~25 MB total)
  char* ws = (char*)d_ws;
  size_t off = 0;
  auto alloc = [&](size_t bytes) { char* p = ws + off; off = (off + bytes + 255) & ~(size_t)255; return p; };
  int*      cnt     = (int*)     alloc((size_t)RN * 4);
  int*      hist    = (int*)     alloc((size_t)R * 4);
  int*      pend    = (int*)     alloc((size_t)(R + 1) * 4);
  int*      tileb   = (int*)     alloc((size_t)(R + 1) * 4);
  int*      cursor  = (int*)     alloc((size_t)R * 4);
  int*      sorted  = (int*)     alloc((size_t)TP * 4);
  float*    nrm     = (float*)   alloc((size_t)T * 4);
  float*    h       = (float*)   alloc((size_t)NE * 4);
  float*    out_pad = (float*)   alloc((size_t)NP * 4);
  _Float16* packedB = (_Float16*)alloc((size_t)R * 4 * 32 * 16 * 2);   // 64 KB
  (void)ws_size;

  const int B = 256;
  int gmax = NP; if (NE > gmax) gmax = NE; if (RN > gmax) gmax = RN; if (TP > gmax) gmax = TP;

  init_kernel<<<(gmax + B - 1) / B, B, 0, stream>>>(cnt, RN, h, NE, out_pad, NP, sorted, TP, hist, R);
  pack_b_kernel<<<(R * 4 * 32 + B - 1) / B, B, 0, stream>>>(w2, packedB, R, C);
  count_kernel<<<(T + B - 1) / B, B, 0, stream>>>(fr, rel, cnt, hist, N, T);
  prefix_kernel<<<1, 32, 0, stream>>>(hist, pend, tileb, cursor, R);
  scatter_kernel<<<(T + B - 1) / B, B, 0, stream>>>(rel, cursor, sorted, T);
  norm_kernel<<<(T + B - 1) / B, B, 0, stream>>>(fr, rel, cnt, nrm, N, T);
  spmm1_kernel<<<((T * 8) + B - 1) / B, B, 0, stream>>>(fr, to, rel, w1, nrm, h, N, T);
  relu_kernel<<<(NE + B - 1) / B, B, 0, stream>>>(h, bias1, NE);

  // upper bound on 32-row supertiles: sum_r ceil(len_r/32) <= T/32 + R
  const int tilesUpper = (T >> 5) + R + 1;
  const int wavesPerBlock = B / 32;
  wmma_out_kernel<<<(tilesUpper + wavesPerBlock - 1) / wavesPerBlock, B, 0, stream>>>(
      sorted, pend, tileb, fr, to, nrm, h, packedB, out_pad, R);

  finalize_kernel<<<(NC + B - 1) / B, B, 0, stream>>>(out_pad, bias2, out, NC, C);
}
